// VQAE_72911364817054
// MI455X (gfx1250) — matrix operand order repaired and checked by reading, not compiled, not run on hardware
//
#include <hip/hip_runtime.h>
#include <hip/hip_bf16.h>

typedef __attribute__((ext_vector_type(16))) _Float16 v16h;
typedef __attribute__((ext_vector_type(8)))  _Float16 v8h;
typedef __attribute__((ext_vector_type(8)))  float    v8f;
typedef __attribute__((ext_vector_type(4)))  unsigned int v4u;
typedef __attribute__((ext_vector_type(8)))  int v8i;
typedef __attribute__((ext_vector_type(4)))  int v4i;

#define D 768
#define K 1024
#define M_TOTAL 16384               // 16 * 1024 rows
#define ROWS_PER_BLOCK 16
#define A_STRIDE 784                 // 768 + 16 halves pad -> conflict-free ds_load_b128
#define N_ELEMS 12582912             // 16*1024*768

#if defined(__has_builtin)
#  if __has_builtin(__builtin_amdgcn_tensor_load_to_lds)
#    define HAVE_TDM 1
#  endif
#endif
#ifndef HAVE_TDM
#  define HAVE_TDM 0
#endif

// ---------------------------------------------------------------------------
// Kernel 1: codebook f32 -> f16, per-code squared norms, zero accumulators
// ---------------------------------------------------------------------------
__global__ void vq_prep_kernel(const float* __restrict__ cb,
                               _Float16* __restrict__ cb16,
                               float* __restrict__ cnorm,
                               int* __restrict__ counts,
                               double* __restrict__ sumsq) {
  __shared__ float red[256];
  const int k = blockIdx.x;          // one block per code row, 1024 blocks
  const int t = threadIdx.x;
  const float* row = cb + (long)k * D;
  float s = 0.0f;
  for (int c = t; c < D; c += 256) {
    float v = row[c];
    cb16[(long)k * D + c] = (_Float16)v;
    s += v * v;
  }
  red[t] = s;
  __syncthreads();
  for (int st = 128; st > 0; st >>= 1) {
    if (t < st) red[t] += red[t + st];
    __syncthreads();
  }
  if (t == 0) cnorm[k] = red[0];
  if (t == 1) counts[k] = 0;
  if (k == 0 && t == 2) *sumsq = 0.0;
}

// ---------------------------------------------------------------------------
// Kernel 2: WMMA distance GEMM + argmin.
// Block = 256 threads (8 waves), owns 16 rows of x. Wave w owns code pairs
// j = w, w+8, w+16, w+24 (each pair = 32 codes = two 16x16 accumulator tiles).
// ---------------------------------------------------------------------------
__global__ void __launch_bounds__(256)
vq_argmin_kernel(const float* __restrict__ x,
                 const _Float16* __restrict__ cb16,
                 const float* __restrict__ cnorm,
                 int* __restrict__ widx) {
  __shared__ alignas(16) _Float16 xs[ROWS_PER_BLOCK * A_STRIDE];
  __shared__ float rmin[8][16];
  __shared__ int   ridx[8][16];

  const int tid = threadIdx.x;
  const long rowbase = (long)blockIdx.x * ROWS_PER_BLOCK;

  // stage x tile -> f16 in LDS (row-major, padded stride)
  const float* xrow = x + rowbase * D;
  for (int p = tid; p < ROWS_PER_BLOCK * D; p += 256) {
    int r = p / D;
    int c = p - r * D;
    xs[r * A_STRIDE + c] = (_Float16)xrow[p];
  }
  __syncthreads();

  const int wave = tid >> 5;
  const int lane = tid & 31;
  const int n    = lane & 15;        // column of C tile / row of A tile
  const int hi   = lane >> 4;        // lane-half selector for A/B K-packing

  float mind[8];
  int   midx[8];
#pragma unroll
  for (int r = 0; r < 8; ++r) { mind[r] = 3.4e38f; midx[r] = 0; }

  for (int j = wave; j < 32; j += 8) {
    const int c0 = j * 32;
    if (j + 8 < 32)                  // hint next pair's B rows toward L2/L0
      __builtin_prefetch(cb16 + (long)((j + 8) * 32 + n) * D, 0, 1);

    const float cn0 = cnorm[c0 + n];
    const float cn1 = cnorm[c0 + 16 + n];
    const _Float16* b0p = cb16 + (long)(c0 + n) * D + hi * 16;
    const _Float16* b1p = b0p + 16 * D;

    v8f acc0 = {};
    v8f acc1 = {};
#pragma unroll 4
    for (int k0 = 0; k0 < D; k0 += 32) {
      // A fragment: 16x32 f16. lanes 0-15: K = k0+[0..15];
      // lanes 16-31: +16.  (two 16B LDS loads)
      const v8h alo = *(const v8h*)&xs[n * A_STRIDE + k0 + hi * 16];
      const v8h ahi = *(const v8h*)&xs[n * A_STRIDE + k0 + hi * 16 + 8];
      const v16h a = __builtin_shufflevector(alo, ahi,
          0, 1, 2, 3, 4, 5, 6, 7, 8, 9, 10, 11, 12, 13, 14, 15);

      // B fragments: 32x16 f16, column n = lane&15, 16 consecutive K halves
      // starting at k0 (+16 for upper lane half): two 16B global loads each.
      const v8h* bq0 = (const v8h*)(b0p + k0);
      const v16h b0 = __builtin_shufflevector(bq0[0], bq0[1],
          0, 1, 2, 3, 4, 5, 6, 7, 8, 9, 10, 11, 12, 13, 14, 15);
      const v8h* bq1 = (const v8h*)(b1p + k0);
      const v16h b1 = __builtin_shufflevector(bq1[0], bq1[1],
          0, 1, 2, 3, 4, 5, 6, 7, 8, 9, 10, 11, 12, 13, 14, 15);

      acc0 = __builtin_amdgcn_wmma_f32_16x16x32_f16(false, a, false, b0,
                                                    (short)0, acc0, false, false);
      acc1 = __builtin_amdgcn_wmma_f32_16x16x32_f16(false, a, false, b1,
                                                    (short)0, acc1, false, false);
    }

    // dist = ||c||^2 - 2 x.c  (||x||^2 constant per row, irrelevant to argmin)
#pragma unroll
    for (int r = 0; r < 8; ++r) {
      float d0 = cn0 - 2.0f * acc0[r];
      float d1 = cn1 - 2.0f * acc1[r];
      int   i0 = c0 + n;
      const int i1 = c0 + 16 + n;
      if (d1 < d0) { d0 = d1; i0 = i1; }
      if (d0 < mind[r] || (d0 == mind[r] && i0 < midx[r])) {
        mind[r] = d0; midx[r] = i0;
      }
    }
  }

  // reduce across the 16 lanes that share each row (wave32 C layout:
  // lanes 0-15 hold rows 0..7 in vgpr r, lanes 16-31 hold rows 8..15)
#pragma unroll
  for (int r = 0; r < 8; ++r) {
#pragma unroll
    for (int off = 8; off >= 1; off >>= 1) {
      float od = __shfl_xor(mind[r], off, 32);
      int   oi = __shfl_xor(midx[r], off, 32);
      if (od < mind[r] || (od == mind[r] && oi < midx[r])) {
        mind[r] = od; midx[r] = oi;
      }
    }
  }
  if (lane == 0) {
#pragma unroll
    for (int r = 0; r < 8; ++r) { rmin[wave][r] = mind[r]; ridx[wave][r] = midx[r]; }
  } else if (lane == 16) {
#pragma unroll
    for (int r = 0; r < 8; ++r) { rmin[wave][8 + r] = mind[r]; ridx[wave][8 + r] = midx[r]; }
  }
  __syncthreads();

  if (tid < 16) {
    float best = rmin[0][tid];
    int   bi   = ridx[0][tid];
#pragma unroll
    for (int w = 1; w < 8; ++w) {
      float d = rmin[w][tid];
      int   i = ridx[w][tid];
      if (d < best || (d == best && i < bi)) { best = d; bi = i; }
    }
    widx[rowbase + tid] = bi;
  }
}

// ---------------------------------------------------------------------------
// Kernel 3: quantized = codebook[idx] via TDM gather-mode (16 row indices in
// D# groups 2/3, one tensor_load_to_lds per block), then stream tile to out
// and accumulate sum((q-x)^2) with an f64 global atomic.
// ---------------------------------------------------------------------------
__global__ void __launch_bounds__(256)
vq_quant_loss_kernel(const float* __restrict__ x,
                     const float* __restrict__ cb,
                     const int* __restrict__ widx,
                     float* __restrict__ out,
                     double* __restrict__ sumsq) {
  __shared__ alignas(16) float qs[ROWS_PER_BLOCK * D];   // 48 KB gathered tile
  __shared__ float red[256];
  const int tid = threadIdx.x;
  const long rowbase = (long)blockIdx.x * ROWS_PER_BLOCK;

#if HAVE_TDM
  if (tid < 32) {                     // one wave issues the DMA (EXEC ignored)
    unsigned int pk[8];
#pragma unroll
    for (int j = 0; j < 8; ++j) {     // uniform addresses -> scalar loads
      unsigned int a = (unsigned int)widx[rowbase + 2 * j];
      unsigned int b = (unsigned int)widx[rowbase + 2 * j + 1];
      pk[j] = a | (b << 16);
    }
    const unsigned long long ga = (unsigned long long)(size_t)cb;
    const unsigned int lds_off = (unsigned int)(size_t)&qs[0];
    // D# group0: count=1, gather_mode=1 (16-bit indices), lds_addr,
    // global_addr[56:0], type=2 in [127:126]
    v4u g0 = { 0x80000001u, lds_off, (unsigned int)ga,
               (unsigned int)((ga >> 32) & 0x01FFFFFFu) | 0x80000000u };
    // D# group1: data_size=4B (enc 2) @[17:16]; tensor_dim0=768 @[79:48];
    // tensor_dim1=1024 @[111:80]; tile_dim0=768 @[127:112];
    // tile_dim1=#indices=16 @[143:128]; tensor_dim0_stride=768 @[207:160]
    v8i g1 = { (int)(2u << 16),
               (int)(768u << 16),
               (int)(1024u << 16),
               (int)(768u << 16),
               16,
               768,
               0, 0 };
    // D# groups 2/3: sixteen 16-bit row indices
    v4i g2 = { (int)pk[0], (int)pk[1], (int)pk[2], (int)pk[3] };
    v4i g3 = { (int)pk[4], (int)pk[5], (int)pk[6], (int)pk[7] };
#if __clang_major__ >= 23
    v8i gz = {};
    __builtin_amdgcn_tensor_load_to_lds(g0, g1, g2, g3, gz, 0);
#else
    __builtin_amdgcn_tensor_load_to_lds(g0, g1, g2, g3, 0);
#endif
    __builtin_amdgcn_s_wait_tensorcnt(0);
  }
#else
  for (int p = tid; p < ROWS_PER_BLOCK * D; p += 256)
    qs[p] = cb[(long)widx[rowbase + p / D] * D + (p % D)];
#endif
  __syncthreads();

  float s = 0.0f;
#pragma unroll
  for (int v = 0; v < 12; ++v) {
    const int p4 = tid + v * 256;               // float4 slot in tile [0,3072)
    const float4 q = *(const float4*)&qs[p4 * 4];
    const long gi = rowbase * D + (long)p4 * 4;
    const float4 xv = *(const float4*)(x + gi);
    *(float4*)(out + gi) = q;
    const float dx = q.x - xv.x, dy = q.y - xv.y, dz = q.z - xv.z, dw = q.w - xv.w;
    s += dx * dx + dy * dy + dz * dz + dw * dw;
  }
  red[tid] = s;
  __syncthreads();
  for (int st = 128; st > 0; st >>= 1) {
    if (tid < st) red[tid] += red[tid + st];
    __syncthreads();
  }
  if (tid == 0) atomicAdd(sumsq, (double)red[0]);
}

// ---------------------------------------------------------------------------
// Kernel 4: code usage histogram
// ---------------------------------------------------------------------------
__global__ void vq_hist_kernel(const int* __restrict__ widx,
                               int* __restrict__ counts) {
  const int m = blockIdx.x * 256 + threadIdx.x;          // 16384 threads
  atomicAdd(&counts[widx[m]], 1);
}

// ---------------------------------------------------------------------------
// Kernel 5: scalars -> loss, perplexity, usage
// ---------------------------------------------------------------------------
__global__ void vq_finalize_kernel(const int* __restrict__ counts,
                                   const double* __restrict__ sumsq,
                                   float* __restrict__ out_tail) {
  __shared__ float entr[1024];
  __shared__ int   used[1024];
  const int k = threadIdx.x;
  const int c = counts[k];
  const float p = (float)c * (1.0f / (float)M_TOTAL);
  entr[k] = -p * logf(p + 1e-10f);
  used[k] = (c > 0) ? 1 : 0;
  __syncthreads();
  for (int st = 512; st > 0; st >>= 1) {
    if (k < st) { entr[k] += entr[k + st]; used[k] += used[k + st]; }
    __syncthreads();
  }
  if (k == 0) {
    // quantizer_loss = 2*(0.25*mse + mse) = 2.5 * mse
    out_tail[0] = (float)(2.5 * (*sumsq) / (double)N_ELEMS);
    out_tail[1] = expf(entr[0]);
    out_tail[2] = (float)used[0] / (float)K;
  }
}

// ---------------------------------------------------------------------------
extern "C" void kernel_launch(void* const* d_in, const int* in_sizes, int n_in,
                              void* d_out, int out_size, void* d_ws, size_t ws_size,
                              hipStream_t stream) {
  const float* x  = (const float*)d_in[0];   // [16,1024,768]
  const float* cb = (const float*)d_in[1];   // [1024,768]
  float* out = (float*)d_out;                // [16,1024,768] ++ 3 scalars

  char* w = (char*)d_ws;
  _Float16* cb16  = (_Float16*)(w);                       // 1,572,864 B
  float*    cnorm = (float*)(w + 1572864);                //     4,096 B
  int*      widx  = (int*)(w + 1576960);                  //    65,536 B
  int*      counts= (int*)(w + 1642496);                  //     4,096 B
  double*   sumsq = (double*)(w + 1646592);               //         8 B

  vq_prep_kernel<<<K, 256, 0, stream>>>(cb, cb16, cnorm, counts, sumsq);
  vq_argmin_kernel<<<M_TOTAL / ROWS_PER_BLOCK, 256, 0, stream>>>(x, cb16, cnorm, widx);
  vq_quant_loss_kernel<<<M_TOTAL / ROWS_PER_BLOCK, 256, 0, stream>>>(x, cb, widx, out, sumsq);
  vq_hist_kernel<<<M_TOTAL / 256, 256, 0, stream>>>(widx, counts);
  vq_finalize_kernel<<<1, 1024, 0, stream>>>(counts, sumsq, out + N_ELEMS);
}
